// GQAWithKVCache_30769145709212
// MI455X (gfx1250) — matrix operand order repaired and checked by reading, not compiled, not run on hardware
//
#include <hip/hip_runtime.h>

// Problem constants (match the reference)
#define S_LEN   2048
#define DMODEL  2048
#define NHQ     32
#define NHKV    8
#define HDIM    64
#define GQA_GRP 4
#define KVDIM   (NHKV * HDIM)   // 512

typedef __bf16 v16bf __attribute__((ext_vector_type(16)));
typedef float  v8f   __attribute__((ext_vector_type(8)));
typedef unsigned short u16x16 __attribute__((ext_vector_type(16)));
typedef unsigned short u16x8  __attribute__((ext_vector_type(8)));
typedef int vsi4 __attribute__((vector_size(16)));   // generic-vector int4 (builtin param type)

union V16 {
  u16x16 u;
  v16bf  bf;
  struct { u16x8 lo; u16x8 hi; } h;
};

__device__ __forceinline__ unsigned short f2bf(float f) {
  unsigned int u = __float_as_uint(f);
  u += 0x7FFFu + ((u >> 16) & 1u);       // round-to-nearest-even
  return (unsigned short)(u >> 16);
}

// ---- CDNA5 async global->LDS staging (ASYNCcnt path) -----------------------
// Each call copies 16 bytes per lane, tracked by ASYNCcnt; falls back to a
// synchronous VGPR round-trip if the builtin is unavailable.
#if defined(__has_builtin)
#if __has_builtin(__builtin_amdgcn_global_load_async_to_lds_b128)
#define HAVE_ASYNC_LDS 1
#endif
#endif

__device__ __forceinline__ void async_copy16(const unsigned short* g,
                                             unsigned short* l) {
#if defined(HAVE_ASYNC_LDS)
  __builtin_amdgcn_global_load_async_to_lds_b128(
      (__attribute__((address_space(1))) vsi4*)(g),
      (__attribute__((address_space(3))) vsi4*)(l), 0, 0);
#else
  *(u16x8*)l = *(const u16x8*)g;
#endif
}

__device__ __forceinline__ void wait_async_all() {
#if defined(HAVE_ASYNC_LDS)
#if __has_builtin(__builtin_amdgcn_s_wait_asynccnt)
  __builtin_amdgcn_s_wait_asynccnt(0);
#else
  asm volatile("s_wait_asynccnt 0" ::: "memory");
#endif
#endif
}

// ---- WMMA operand loaders (CDNA5 16-bit layouts, wave32) -------------------
// A (16xK tile, M = lane%16): lanes 0-15 hold K = [kk+0..7] then [kk+16..23];
// lanes 16-31 hold K = [kk+8..15] then [kk+24..31]  -> two 8-elem runs.
__device__ __forceinline__ v16bf load_a16(const unsigned short* A, int lda,
                                          int m0, int kk, int lane) {
  const int m = m0 + (lane & 15);
  const int half8 = (lane >> 4) << 3;
  const unsigned short* p = A + (long)m * lda + kk + half8;
  V16 r;
  r.h.lo = *(const u16x8*)(p);
  r.h.hi = *(const u16x8*)(p + 16);
  return r.bf;
}

// B (Kx16 tile, N = lane%16): lanes 0-15 hold K = [kk+0..7] then [kk+16..23];
// lanes 16-31 hold K = [kk+8..15] then [kk+24..31]. Bt is stored N-major
// (Bt[n*ldb + k]) so this is two 16-byte loads per lane.
__device__ __forceinline__ v16bf load_b16(const unsigned short* Bt, int ldb,
                                          int n0, int kk, int lane) {
  const int n = n0 + (lane & 15);
  const int kb = kk + ((lane >> 4) << 3);
  const unsigned short* p = Bt + (long)n * ldb + kb;
  V16 r;
  r.h.lo = *(const u16x8*)(p);
  r.h.hi = *(const u16x8*)(p + 16);
  return r.bf;
}

// ---- RMSNorm + cast to bf16 ------------------------------------------------
__global__ __launch_bounds__(256)
void rmsnorm_kernel(const float* __restrict__ x, const float* __restrict__ g,
                    unsigned short* __restrict__ h) {
  const int row = blockIdx.x;
  const float* xr = x + (long)row * DMODEL;
  float ss = 0.f;
  for (int i = threadIdx.x; i < DMODEL; i += blockDim.x) {
    float v = xr[i];
    ss += v * v;
  }
  for (int off = 16; off > 0; off >>= 1) ss += __shfl_xor(ss, off);
  __shared__ float red[8];
  __shared__ float total;
  if ((threadIdx.x & 31) == 0) red[threadIdx.x >> 5] = ss;
  __syncthreads();
  if (threadIdx.x == 0) {
    float t = 0.f;
    for (int i = 0; i < 8; ++i) t += red[i];
    total = t;
  }
  __syncthreads();
  const float scale = rsqrtf(total / (float)DMODEL + 1e-9f);
  for (int i = threadIdx.x; i < DMODEL; i += blockDim.x)
    h[(long)row * DMODEL + i] = f2bf(xr[i] * scale * g[i]);
}

// ---- transpose + cast fp32 weight [K][N] -> bf16 [N][K] --------------------
__global__ __launch_bounds__(256)
void transpose_cast_kernel(const float* __restrict__ w,
                           unsigned short* __restrict__ wt, int K, int N) {
  long idx = (long)blockIdx.x * blockDim.x + threadIdx.x;
  if (idx >= (long)K * N) return;
  int n = (int)(idx % N);
  int k = (int)(idx / N);
  wt[(long)n * K + k] = f2bf(w[idx]);
}

// ---- LDS-tiled bf16 WMMA GEMM: C = A(MxK) * Bt(NxK)^T ----------------------
// 64x64 macro-tile per 256-thread block; A/B k-slabs (64x32 bf16 = 4KB each)
// double-buffered in LDS via async global->LDS copies. Each of the 8 waves
// computes a 16x32 sub-tile (1 A-frag, 2 B-frags, 2 WMMAs per k-step).
// MODE 1: fp32 row-major out   MODE 2: bf16 transposed out (Cout[n*ldo+m])
// MODE 3: fp32 row-major + residual
template <int MODE>
__global__ __launch_bounds__(256)
void wmma_gemm_lds_kernel(const unsigned short* __restrict__ A,
                          const unsigned short* __restrict__ Bt,
                          void* __restrict__ Cout,
                          const float* __restrict__ residual,
                          int M, int N, int K, int ldo) {
  __shared__ unsigned short sA[2][64 * 32];
  __shared__ unsigned short sB[2][64 * 32];

  const int tid  = threadIdx.x;
  const int lane = tid & 31;
  const int wv   = tid >> 5;          // 0..7
  const int tilesN = N >> 6;
  const int m0 = (blockIdx.x / tilesN) << 6;
  const int n0 = (blockIdx.x % tilesN) << 6;

  // staging: each thread owns one 16B segment of the A slab and the B slab
  const int row  = tid >> 2;          // 0..63
  const int seg  = (tid & 3) << 3;    // 0,8,16,24 elements
  const unsigned short* gA = A  + (long)(m0 + row) * K + seg;
  const unsigned short* gB = Bt + (long)(n0 + row) * K + seg;
  const int lofs = row * 32 + seg;

  // wave sub-tile assignment inside the 64x64 macro tile
  const int tm  = (wv & 3) << 4;        // m offset: 0,16,32,48
  const int tn0 = (wv >> 2) << 5;       // n offset: 0 or 32 (covers 2 tiles)

  // prestage k-slab 0
  async_copy16(gA, &sA[0][lofs]);
  async_copy16(gB, &sB[0][lofs]);

  v8f acc0 = {}, acc1 = {};
  int buf = 0;
  for (int kk = 0; kk < K; kk += 32) {
    wait_async_all();        // our slab copies for `buf` have landed
    __syncthreads();         // everyone's copies visible; prev compute done
    if (kk + 32 < K) {       // issue next slab into the other buffer
      async_copy16(gA + kk + 32, &sA[buf ^ 1][lofs]);
      async_copy16(gB + kk + 32, &sB[buf ^ 1][lofs]);
    }
    v16bf a  = load_a16(sA[buf], 32, tm, 0, lane);
    v16bf b0 = load_b16(sB[buf], 32, tn0, 0, lane);
    v16bf b1 = load_b16(sB[buf], 32, tn0 + 16, 0, lane);
    acc0 = __builtin_amdgcn_wmma_f32_16x16x32_bf16(false, a, false, b0,
                                                   (short)0, acc0, false, false);
    acc1 = __builtin_amdgcn_wmma_f32_16x16x32_bf16(false, a, false, b1,
                                                   (short)0, acc1, false, false);
    buf ^= 1;
  }

  // C layout: lane holds N = lane%16, rows M = 8*(lane/16) + r
  const int mr  = m0 + tm + ((lane >> 4) << 3);
  const int nc0 = n0 + tn0 + (lane & 15);
#pragma unroll
  for (int r = 0; r < 8; ++r) {
    const int m = mr + r;
    if (MODE == 1) {
      ((float*)Cout)[(long)m * ldo + nc0]      = acc0[r];
      ((float*)Cout)[(long)m * ldo + nc0 + 16] = acc1[r];
    } else if (MODE == 2) {
      ((unsigned short*)Cout)[(long)nc0 * ldo + m]        = f2bf(acc0[r]);
      ((unsigned short*)Cout)[(long)(nc0 + 16) * ldo + m] = f2bf(acc1[r]);
    } else {
      ((float*)Cout)[(long)m * ldo + nc0] =
          acc0[r] + residual[(long)m * ldo + nc0];
      ((float*)Cout)[(long)m * ldo + nc0 + 16] =
          acc1[r] + residual[(long)m * ldo + nc0 + 16];
    }
  }
}

// ---- RoPE (rotate-half), fp32 in -> bf16 out -------------------------------
__global__ __launch_bounds__(256)
void rope_kernel(const float* __restrict__ in, unsigned short* __restrict__ out,
                 int H) {
  const int half = HDIM / 2;  // 32
  int idx = blockIdx.x * blockDim.x + threadIdx.x;
  if (idx >= S_LEN * H * half) return;
  const int d = idx % half;
  const int h = (idx / half) % H;
  const int s = idx / (half * H);
  // inv_freq = 10000^(-(2d)/64)
  const float inv_freq = __expf(-(float)(2 * d) * (9.210340371976184f / (float)HDIM));
  const float ang = (float)s * inv_freq;
  float sn, cs;
  __sincosf(ang, &sn, &cs);
  const float* base = in + ((long)s * H + h) * HDIM;
  unsigned short* ob = out + ((long)s * H + h) * HDIM;
  const float x1 = base[d], x2 = base[d + half];
  ob[d]        = f2bf(x1 * cs - x2 * sn);
  ob[d + half] = f2bf(x1 * sn + x2 * cs);
}

// ---- causal GQA flash attention: one wave per (16-row q tile, q head) ------
__global__ __launch_bounds__(256)
void flash_attn_kernel(const unsigned short* __restrict__ qb,   // [S][HQ*HD] bf16
                       const unsigned short* __restrict__ kb,   // [S][HKV*HD] bf16
                       const unsigned short* __restrict__ vt,   // [HKV*HD][S] bf16
                       unsigned short* __restrict__ ob) {       // [S][HQ*HD] bf16
  __shared__ unsigned short lds_p[8][16 * 32];   // per-wave P tile scratch
  const int lane  = threadIdx.x & 31;
  const int wslot = threadIdx.x >> 5;
  const int wid   = blockIdx.x * 8 + wslot;
  const int h  = wid % NHQ;
  const int qt = wid / NHQ;
  if (qt >= S_LEN / 16) return;
  const int q0  = qt << 4;
  const int kvh = h / GQA_GRP;
  const float scale = 0.125f;  // 1/sqrt(64)

  float m_r[8], l_r[8];
  v8f o0 = {}, o1 = {}, o2 = {}, o3 = {};
#pragma unroll
  for (int r = 0; r < 8; ++r) { m_r[r] = -1e30f; l_r[r] = 0.f; }

  const unsigned short* kbase = kb + kvh * HDIM;                // n-major, ld = KVDIM
  const unsigned short* vbase = vt + (long)kvh * HDIM * S_LEN;  // n-major, ld = S
  unsigned short* lp = lds_p[wslot];

  const int kv_end = q0 + 16;
  for (int kv0 = 0; kv0 < kv_end; kv0 += 32) {
    // ---- scores S = Q K^T : two 16x16 tiles over 32 kv positions ----
    v8f s0 = {}, s1 = {};
#pragma unroll
    for (int kk = 0; kk < HDIM; kk += 32) {
      v16bf a  = load_a16(qb + h * HDIM, NHQ * HDIM, q0, kk, lane);
      v16bf b0 = load_b16(kbase, KVDIM, kv0,      kk, lane);
      v16bf b1 = load_b16(kbase, KVDIM, kv0 + 16, kk, lane);
      s0 = __builtin_amdgcn_wmma_f32_16x16x32_bf16(false, a, false, b0,
                                                   (short)0, s0, false, false);
      s1 = __builtin_amdgcn_wmma_f32_16x16x32_bf16(false, a, false, b1,
                                                   (short)0, s1, false, false);
    }
    // ---- causal mask + scale + online softmax ----
    const int mrow = q0 + ((lane >> 4) << 3);
    const int ng   = kv0 + (lane & 15);
    float alpha[8];
#pragma unroll
    for (int r = 0; r < 8; ++r) {
      const int m = mrow + r;
      float v0 = (ng      <= m) ? s0[r] * scale : -1e30f;
      float v1 = (ng + 16 <= m) ? s1[r] * scale : -1e30f;
      float mx = fmaxf(v0, v1);
      for (int off = 1; off < 16; off <<= 1) mx = fmaxf(mx, __shfl_xor(mx, off));
      const float mnew = fmaxf(m_r[r], mx);
      alpha[r] = __expf(m_r[r] - mnew);
      m_r[r] = mnew;
      const float p0 = __expf(v0 - mnew);
      const float p1 = __expf(v1 - mnew);
      s0[r] = p0; s1[r] = p1;
      float rs = p0 + p1;
      for (int off = 1; off < 16; off <<= 1) rs += __shfl_xor(rs, off);
      l_r[r] = l_r[r] * alpha[r] + rs;
      o0[r] *= alpha[r]; o1[r] *= alpha[r];
      o2[r] *= alpha[r]; o3[r] *= alpha[r];
    }
    // ---- re-layout P: C-tile -> LDS -> A-tile (bf16) ----
#pragma unroll
    for (int r = 0; r < 8; ++r) {
      const int M = ((lane >> 4) << 3) + r;
      const int N = lane & 15;
      lp[M * 32 + N]      = f2bf(s0[r]);
      lp[M * 32 + 16 + N] = f2bf(s1[r]);
    }
    const int Ma = lane & 15;
    const int hh = (lane >> 4) << 3;
    V16 pa;
    pa.h.lo = *(const u16x8*)(lp + Ma * 32 + hh);
    pa.h.hi = *(const u16x8*)(lp + Ma * 32 + 16 + hh);
    // ---- O += P * V (4 d-chunks of 16) ----
    o0 = __builtin_amdgcn_wmma_f32_16x16x32_bf16(false, pa.bf, false,
           load_b16(vbase, S_LEN,  0, kv0, lane), (short)0, o0, false, false);
    o1 = __builtin_amdgcn_wmma_f32_16x16x32_bf16(false, pa.bf, false,
           load_b16(vbase, S_LEN, 16, kv0, lane), (short)0, o1, false, false);
    o2 = __builtin_amdgcn_wmma_f32_16x16x32_bf16(false, pa.bf, false,
           load_b16(vbase, S_LEN, 32, kv0, lane), (short)0, o2, false, false);
    o3 = __builtin_amdgcn_wmma_f32_16x16x32_bf16(false, pa.bf, false,
           load_b16(vbase, S_LEN, 48, kv0, lane), (short)0, o3, false, false);
  }
  // ---- normalize + store bf16 attention output ----
#pragma unroll
  for (int r = 0; r < 8; ++r) {
    const float inv = 1.f / l_r[r];
    const int m = q0 + ((lane >> 4) << 3) + r;
    const int n = lane & 15;
    unsigned short* orow = ob + (long)m * DMODEL + h * HDIM;
    orow[n]      = f2bf(o0[r] * inv);
    orow[16 + n] = f2bf(o1[r] * inv);
    orow[32 + n] = f2bf(o2[r] * inv);
    orow[48 + n] = f2bf(o3[r] * inv);
  }
}

// ---------------------------------------------------------------------------
extern "C" void kernel_launch(void* const* d_in, const int* in_sizes, int n_in,
                              void* d_out, int out_size, void* d_ws,
                              size_t ws_size, hipStream_t stream) {
  const float* x  = (const float*)d_in[0];
  const float* g  = (const float*)d_in[1];
  const float* wq = (const float*)d_in[2];
  const float* wk = (const float*)d_in[3];
  const float* wv = (const float*)d_in[4];
  const float* wo = (const float*)d_in[5];
  float* out = (float*)d_out;

  // workspace carving (bytes)
  char* ws = (char*)d_ws;
  size_t off = 0;
  auto carve = [&](size_t bytes) { void* p = ws + off; off += (bytes + 255) & ~(size_t)255; return p; };
  unsigned short* h_bf  = (unsigned short*)carve((size_t)S_LEN * DMODEL * 2);
  unsigned short* wq_t  = (unsigned short*)carve((size_t)DMODEL * DMODEL * 2);
  unsigned short* wk_t  = (unsigned short*)carve((size_t)DMODEL * KVDIM * 2);
  unsigned short* wv_t  = (unsigned short*)carve((size_t)DMODEL * KVDIM * 2);
  unsigned short* wo_t  = (unsigned short*)carve((size_t)DMODEL * DMODEL * 2);
  float*          q_f   = (float*)carve((size_t)S_LEN * DMODEL * 4);
  float*          k_f   = (float*)carve((size_t)S_LEN * KVDIM * 4);
  unsigned short* q_bf  = (unsigned short*)carve((size_t)S_LEN * DMODEL * 2);
  unsigned short* k_bf  = (unsigned short*)carve((size_t)S_LEN * KVDIM * 2);
  unsigned short* vt_bf = (unsigned short*)carve((size_t)KVDIM * S_LEN * 2);
  unsigned short* at_bf = (unsigned short*)carve((size_t)S_LEN * DMODEL * 2);

  // 1) RMSNorm -> bf16 activations
  rmsnorm_kernel<<<S_LEN, 256, 0, stream>>>(x, g, h_bf);

  // 2) weight transpose+cast to bf16 N-major
  transpose_cast_kernel<<<(DMODEL * DMODEL + 255) / 256, 256, 0, stream>>>(wq, wq_t, DMODEL, DMODEL);
  transpose_cast_kernel<<<(DMODEL * KVDIM + 255) / 256, 256, 0, stream>>>(wk, wk_t, DMODEL, KVDIM);
  transpose_cast_kernel<<<(DMODEL * KVDIM + 255) / 256, 256, 0, stream>>>(wv, wv_t, DMODEL, KVDIM);
  transpose_cast_kernel<<<(DMODEL * DMODEL + 255) / 256, 256, 0, stream>>>(wo, wo_t, DMODEL, DMODEL);

  // 3) projections (WMMA, LDS-tiled): Q,K -> fp32 (for RoPE), V -> bf16 transposed
  wmma_gemm_lds_kernel<1><<<(S_LEN / 64) * (DMODEL / 64), 256, 0, stream>>>(
      h_bf, wq_t, q_f, nullptr, S_LEN, DMODEL, DMODEL, DMODEL);
  wmma_gemm_lds_kernel<1><<<(S_LEN / 64) * (KVDIM / 64), 256, 0, stream>>>(
      h_bf, wk_t, k_f, nullptr, S_LEN, KVDIM, DMODEL, KVDIM);
  wmma_gemm_lds_kernel<2><<<(S_LEN / 64) * (KVDIM / 64), 256, 0, stream>>>(
      h_bf, wv_t, vt_bf, nullptr, S_LEN, KVDIM, DMODEL, S_LEN);

  // 4) RoPE -> bf16 Q/K
  rope_kernel<<<(S_LEN * NHQ * (HDIM / 2) + 255) / 256, 256, 0, stream>>>(q_f, q_bf, NHQ);
  rope_kernel<<<(S_LEN * NHKV * (HDIM / 2) + 255) / 256, 256, 0, stream>>>(k_f, k_bf, NHKV);

  // 5) causal GQA flash attention (WMMA)
  flash_attn_kernel<<<(S_LEN / 16) * NHQ / 8, 256, 0, stream>>>(q_bf, k_bf, vt_bf, at_bf);

  // 6) output projection + residual (WMMA, LDS-tiled), fp32 out
  wmma_gemm_lds_kernel<3><<<(S_LEN / 64) * (DMODEL / 64), 256, 0, stream>>>(
      at_bf, wo_t, out, x, S_LEN, DMODEL, DMODEL, DMODEL);

  (void)in_sizes; (void)n_in; (void)out_size; (void)ws_size;
}
